// LSTM_Attention_66898410602709
// MI455X (gfx1250) — matrix operand order repaired and checked by reading, not compiled, not run on hardware
//
#include <hip/hip_runtime.h>
#include <hip/hip_bf16.h>
#include <stdint.h>

// ---------------------------------------------------------------------------
// Seq2seq LSTM+attention for MI455X (gfx1250).
// Per-step fused [x_t ; h] * [Wih | Whh]^T via v_wmma_f32_16x16x32_bf16,
// f32 accumulation, f32 cell state. Weights packed to bf16 once per call
// (~24MB, resident in the 192MB L2 across the whole scan).
// GEMM: 128-thread blocks, block tile 128x64, wave tile 32x64
//   -> 8 WMMA per k-step, 12 b128 loads per k-step (1.5 loads/WMMA).
// ---------------------------------------------------------------------------

constexpr int BB = 256, SS = 256, FF = 64, HE = 512, HD = 512, NOUT = 3;
constexpr int G4 = 4 * HE;   // 2048 gate width
constexpr int EO = 2 * HE;   // 1024 (bi-dir encoder output)

typedef __bf16 bhalf;
typedef __attribute__((ext_vector_type(16))) bhalf        v16bf;
typedef __attribute__((ext_vector_type(8)))  float        v8f;
typedef __attribute__((ext_vector_type(4)))  unsigned int v4u;

union Frag16 { v16bf h; v4u u[2]; };

__device__ __forceinline__ unsigned short f2bf(float x) {
  unsigned int u = __float_as_uint(x);
  u += 0x7FFFu + ((u >> 16) & 1u);           // round-to-nearest-even
  return (unsigned short)(u >> 16);
}
__device__ __forceinline__ float bf2f(unsigned short h) {
  return __uint_as_float(((unsigned int)h) << 16);
}
__device__ __forceinline__ float sigf(float x) { return 1.0f / (1.0f + __expf(-x)); }

// --------------------------- f32 -> bf16 convert ---------------------------
__global__ void k_cvt_bf16(const float* __restrict__ src,
                           unsigned short* __restrict__ dst, int n) {
  int i = blockIdx.x * blockDim.x + threadIdx.x;
  if (i < n) dst[i] = f2bf(src[i]);
}

// ----------------- pack [Wih | Whh] -> bf16, b = bih + bhh -----------------
__global__ __launch_bounds__(128)
void k_packw(const float* __restrict__ wih, const float* __restrict__ whh,
             const float* __restrict__ bih, const float* __restrict__ bhh,
             int din, unsigned short* __restrict__ wc, float* __restrict__ bc) {
  const int nrow = blockIdx.x;            // 0..2047 gate row
  const int K = din + HE;
  for (int k = threadIdx.x; k < K; k += 128) {
    float v = (k < din) ? wih[(size_t)nrow * din + k]
                        : whh[(size_t)nrow * HE + (k - din)];
    wc[(size_t)nrow * K + k] = f2bf(v);
  }
  if (threadIdx.x == 0) bc[nrow] = bih[nrow] + bhh[nrow];
}

// ------------------- per-step fused gate GEMM (WMMA bf16) ------------------
// g[dir] = [x_s ; h_dir] (B x K) * Wcomb_dir^T (K x 2048) + bcomb_dir
// block = 128 threads (4 waves); block tile 128x64; wave tile 32x64.
// gridDim.z = #directions; bwd direction processes s = SS-1-t.
__global__ __launch_bounds__(128)
void k_lstm_gemm(const unsigned short* __restrict__ xbase, int din, int xbstride,
                 const unsigned short* __restrict__ hbf, int hdirstride,
                 const unsigned short* __restrict__ wc, int wdirstride,
                 const float* __restrict__ bc,
                 float* __restrict__ gbuf,
                 int t, int rev_bwd) {
  const int K   = din + HE;
  const int dir = blockIdx.z;
  const int s   = (dir && rev_bwd) ? (SS - 1 - t) : t;

  const unsigned short* xstep = xbase + (size_t)s * din;
  const unsigned short* hp    = hbf + (size_t)dir * hdirstride;
  const unsigned short* wp    = wc  + (size_t)dir * wdirstride;
  const float*          bp    = bc  + dir * G4;
  float*                gout  = gbuf + (size_t)dir * ((size_t)BB * G4);

  const int lane = threadIdx.x & 31;
  const int wave = threadIdx.x >> 5;
  const int half = lane >> 4;        // which 16-lane half of the wave
  const int l15  = lane & 15;
  const int m0   = blockIdx.y * 128 + wave * 32;  // output row base (batch)
  const int n0   = blockIdx.x * 64;               // output col base (gate)

  v8f acc[2][4] = {};                // 2 row-tiles x 4 col-tiles of 16x16 f32

  for (int k0 = 0; k0 < K; k0 += 32) {
    // ---- A fragments: 16x32 bf16 per ISA layout; lane = row (m? + l15),
    //      K chunks {half*8..+7} and {16+half*8..+7}  (two b128 loads each).
    //      din and HE are multiples of 32, so a k-tile never straddles x|h.
    const unsigned short* abase;
    size_t rstride;
    int kk;
    if (k0 < din) { abase = xstep; rstride = (size_t)xbstride; kk = k0; }
    else          { abase = hp;    rstride = HE;               kk = k0 - din; }
    Frag16 a[2];
#pragma unroll
    for (int i = 0; i < 2; ++i) {
      const unsigned short* ap =
          abase + (size_t)(m0 + i * 16 + l15) * rstride + (kk + half * 8);
      a[i].u[0] = *(const v4u*)(ap);
      a[i].u[1] = *(const v4u*)(ap + 16);
    }

#pragma unroll
    for (int j = 0; j < 4; ++j) {
      // ---- B fragment: 32x16; lane = column (n0+j*16+l15), K chunks
      //      {half*8..+7} and {16+half*8..+7} of the packed weight row (two b128 loads).
      //      Each B fragment feeds two back-to-back WMMAs (rows i=0,1).
      const unsigned short* bq =
          wp + (size_t)(n0 + j * 16 + l15) * K + (k0 + half * 8);
      Frag16 b;
      b.u[0] = *(const v4u*)(bq);
      b.u[1] = *(const v4u*)(bq + 16);
      acc[0][j] = __builtin_amdgcn_wmma_f32_16x16x32_bf16(
          false, a[0].h, false, b.h, (short)0, acc[0][j], false, false);
      acc[1][j] = __builtin_amdgcn_wmma_f32_16x16x32_bf16(
          false, a[1].h, false, b.h, (short)0, acc[1][j], false, false);
    }
  }

  // ---- epilogue: C layout = lane col l15, rows r + 8*half; add bias
#pragma unroll
  for (int j = 0; j < 4; ++j) {
    const int col   = n0 + j * 16 + l15;
    const float bia = bp[col];
#pragma unroll
    for (int i = 0; i < 2; ++i) {
      float* gp = gout + (size_t)(m0 + i * 16 + half * 8) * G4 + col;
#pragma unroll
      for (int r = 0; r < 8; ++r) gp[(size_t)r * G4] = acc[i][j][r] + bia;
    }
  }
}

// -------------------- per-step cell update (elementwise) -------------------
__global__ __launch_bounds__(256)
void k_lstm_update(const float* __restrict__ gbuf,
                   float* __restrict__ hstf, float* __restrict__ cstf,
                   unsigned short* __restrict__ hstb, int dirstride,
                   unsigned short* __restrict__ outseq, int out_ld,
                   int col_dirstride, int t, int rev_bwd) {
  const int dir = blockIdx.y;
  const int idx = blockIdx.x * 256 + threadIdx.x;    // 0 .. B*512-1
  const int b = idx >> 9, nn = idx & 511;
  const float* g = gbuf + (size_t)dir * ((size_t)BB * G4) + (size_t)b * G4;
  const float gi = g[nn], gf = g[HE + nn], gg = g[2 * HE + nn], go = g[3 * HE + nn];
  const size_t so = (size_t)dir * dirstride + idx;
  float c = cstf[so];
  c = sigf(gf) * c + sigf(gi) * tanhf(gg);
  const float h = sigf(go) * tanhf(c);
  cstf[so] = c;
  hstf[so] = h;
  const unsigned short hb = f2bf(h);
  hstb[so] = hb;
  const int s = (dir && rev_bwd) ? (SS - 1 - t) : t;
  outseq[(size_t)(b * SS + s) * out_ld + dir * col_dirstride + nn] = hb;
}

// ------------------------------ attention ----------------------------------
__global__ __launch_bounds__(128)
void k_decterm(const float* __restrict__ dh0, const float* __restrict__ aW,
               float* __restrict__ decterm) {
  __shared__ float red[128];
  const int b = blockIdx.x;
  float p = 0.f;
  for (int k = threadIdx.x; k < HD; k += 128) p += dh0[(size_t)b * HD + k] * aW[k];
  red[threadIdx.x] = p; __syncthreads();
  for (int off = 64; off > 0; off >>= 1) {
    if (threadIdx.x < off) red[threadIdx.x] += red[threadIdx.x + off];
    __syncthreads();
  }
  if (threadIdx.x == 0) decterm[b] = red[0];
}

__global__ __launch_bounds__(128)
void k_scores(const unsigned short* __restrict__ encout,
              const float* __restrict__ aW, const float* __restrict__ ab,
              const float* __restrict__ decterm, float* __restrict__ scores) {
  __shared__ float red[128];
  const int idx = blockIdx.x;            // b*SS + s
  const int b = idx / SS;
  const unsigned short* e = encout + (size_t)idx * EO;
  const float* we = aW + HD;             // W_enc = aW[0, HD:]
  float p = 0.f;
  for (int k = threadIdx.x; k < EO; k += 128) p += bf2f(e[k]) * we[k];
  red[threadIdx.x] = p; __syncthreads();
  for (int off = 64; off > 0; off >>= 1) {
    if (threadIdx.x < off) red[threadIdx.x] += red[threadIdx.x + off];
    __syncthreads();
  }
  if (threadIdx.x == 0) scores[idx] = red[0] + decterm[b] + ab[0];
}

__global__ __launch_bounds__(256)
void k_softmax(const float* __restrict__ scores, float* __restrict__ alphaw,
               float* __restrict__ alpha_out) {
  __shared__ float red[256];
  const int b = blockIdx.x, t = threadIdx.x;
  const float v = scores[(size_t)b * SS + t];
  red[t] = v; __syncthreads();
  for (int off = 128; off > 0; off >>= 1) {
    if (t < off) red[t] = fmaxf(red[t], red[t + off]);
    __syncthreads();
  }
  const float m = red[0]; __syncthreads();
  const float e = __expf(v - m);
  red[t] = e; __syncthreads();
  for (int off = 128; off > 0; off >>= 1) {
    if (t < off) red[t] += red[t + off];
    __syncthreads();
  }
  const float a = e / red[0];
  alphaw[(size_t)b * SS + t]    = a;
  alpha_out[(size_t)b * SS + t] = a;
}

__global__ __launch_bounds__(128)
void k_build_decin(const float* __restrict__ alphaw,
                   const unsigned short* __restrict__ encout,
                   const unsigned short* __restrict__ xbf,
                   unsigned short* __restrict__ decin) {
  const int idx = blockIdx.x;            // b*SS + s
  const float a = alphaw[idx];
  const unsigned short* e  = encout + (size_t)idx * EO;
  const unsigned short* xb = xbf    + (size_t)idx * FF;
  unsigned short* d = decin + (size_t)idx * (EO + FF);
  for (int k = threadIdx.x; k < EO; k += 128) d[k] = f2bf(a * bf2f(e[k]));
  if (threadIdx.x < FF) d[EO + threadIdx.x] = xb[threadIdx.x];
}

// --------------------------------- FC head ---------------------------------
__global__ __launch_bounds__(256)
void k_fc1(const unsigned short* __restrict__ decout,
           const float* __restrict__ f1W, const float* __restrict__ f1b,
           float* __restrict__ h1) {
  __shared__ float red[256];
  const int j = blockIdx.x, b = blockIdx.y;
  const unsigned short* dsrc = decout + (size_t)b * ((size_t)SS * HD);
  const float* w = f1W + (size_t)j * ((size_t)SS * HD);
  float p = 0.f;
  for (int k = threadIdx.x; k < SS * HD; k += 256) p += bf2f(dsrc[k]) * w[k];
  red[threadIdx.x] = p; __syncthreads();
  for (int off = 128; off > 0; off >>= 1) {
    if (threadIdx.x < off) red[threadIdx.x] += red[threadIdx.x + off];
    __syncthreads();
  }
  if (threadIdx.x == 0) h1[b * 8 + j] = fmaxf(red[0] + f1b[j], 0.f);
}

__global__ __launch_bounds__(256)
void k_out(const float* __restrict__ h1, const float* __restrict__ f2W,
           const float* __restrict__ f2b, float* __restrict__ out) {
  const int b = threadIdx.x;
  float lg[NOUT];
  float m = -1e30f;
#pragma unroll
  for (int o = 0; o < NOUT; ++o) {
    float s = f2b[o];
#pragma unroll
    for (int k = 0; k < 8; ++k) s += h1[b * 8 + k] * f2W[o * 8 + k];
    lg[o] = s; m = fmaxf(m, s);
  }
  float sum = 0.f;
#pragma unroll
  for (int o = 0; o < NOUT; ++o) { lg[o] = __expf(lg[o] - m); sum += lg[o]; }
#pragma unroll
  for (int o = 0; o < NOUT; ++o) out[(size_t)b * NOUT + o] = lg[o] / sum;
}

// ================================ launcher =================================
extern "C" void kernel_launch(void* const* d_in, const int* in_sizes, int n_in,
                              void* d_out, int out_size, void* d_ws, size_t ws_size,
                              hipStream_t stream) {
  (void)in_sizes; (void)n_in; (void)out_size; (void)ws_size;
  const float* x      = (const float*)d_in[0];
  const float* enc_h0 = (const float*)d_in[1];
  const float* enc_c0 = (const float*)d_in[2];
  const float* dec_h0 = (const float*)d_in[3];
  const float* dec_c0 = (const float*)d_in[4];
  const float* eW0f = (const float*)d_in[5];  const float* eU0f = (const float*)d_in[6];
  const float* eb0f = (const float*)d_in[7];  const float* ec0f = (const float*)d_in[8];
  const float* eW0b = (const float*)d_in[9];  const float* eU0b = (const float*)d_in[10];
  const float* eb0b = (const float*)d_in[11]; const float* ec0b = (const float*)d_in[12];
  const float* eW1f = (const float*)d_in[13]; const float* eU1f = (const float*)d_in[14];
  const float* eb1f = (const float*)d_in[15]; const float* ec1f = (const float*)d_in[16];
  const float* eW1b = (const float*)d_in[17]; const float* eU1b = (const float*)d_in[18];
  const float* eb1b = (const float*)d_in[19]; const float* ec1b = (const float*)d_in[20];
  const float* dW  = (const float*)d_in[21];  const float* dU  = (const float*)d_in[22];
  const float* db  = (const float*)d_in[23];  const float* dcb = (const float*)d_in[24];
  const float* aW  = (const float*)d_in[25];  const float* ab  = (const float*)d_in[26];
  const float* f1W = (const float*)d_in[27];  const float* f1b = (const float*)d_in[28];
  const float* f2W = (const float*)d_in[29];  const float* f2b = (const float*)d_in[30];

  // ---- carve workspace (all offsets 256B aligned) ----
  char* p = (char*)d_ws;
  auto carve = [&](size_t bytes) -> void* {
    void* r = (void*)p; p += (bytes + 255) & ~(size_t)255; return r;
  };
  unsigned short* xbf    = (unsigned short*)carve((size_t)BB * SS * FF * 2);
  unsigned short* inp1   = (unsigned short*)carve((size_t)BB * SS * EO * 2);
  unsigned short* encout = (unsigned short*)carve((size_t)BB * SS * EO * 2);
  unsigned short* decin  = (unsigned short*)carve((size_t)BB * SS * (EO + FF) * 2);
  unsigned short* decout = (unsigned short*)carve((size_t)BB * SS * HD * 2);
  float* gbuf  = (float*)carve((size_t)2 * BB * G4 * 4);
  float* hstf  = (float*)carve((size_t)4 * BB * HE * 4);   // [f0,b0,f1,b1]
  float* cstf  = (float*)carve((size_t)4 * BB * HE * 4);
  unsigned short* hstb = (unsigned short*)carve((size_t)4 * BB * HE * 2);
  float* dhf = (float*)carve((size_t)BB * HD * 4);
  float* dcf = (float*)carve((size_t)BB * HD * 4);
  unsigned short* dhb = (unsigned short*)carve((size_t)BB * HD * 2);
  unsigned short* wc_e0 = (unsigned short*)carve((size_t)2 * G4 * (FF + HE) * 2);
  unsigned short* wc_e1 = (unsigned short*)carve((size_t)2 * G4 * (EO + HE) * 2);
  unsigned short* wc_d  = (unsigned short*)carve((size_t)G4 * (EO + FF + HE) * 2);
  float* bc      = (float*)carve((size_t)5 * G4 * 4);
  float* scores  = (float*)carve((size_t)BB * SS * 4);
  float* alphaw  = (float*)carve((size_t)BB * SS * 4);
  float* decterm = (float*)carve((size_t)BB * 4);
  float* h1      = (float*)carve((size_t)BB * 8 * 4);

  // ---- output layout (concat in reference return order) ----
  float* out       = (float*)d_out;
  float* out_hs    = out + BB * NOUT;
  float* out_cs    = out_hs + 4 * BB * HE;
  float* out_dh    = out_cs + 4 * BB * HE;
  float* out_dc    = out_dh + BB * HD;
  float* out_alpha = out_dc + BB * HD;

  // ---- init states + bf16 conversions ----
  hipMemcpyAsync(hstf, enc_h0, (size_t)4 * BB * HE * 4, hipMemcpyDeviceToDevice, stream);
  hipMemcpyAsync(cstf, enc_c0, (size_t)4 * BB * HE * 4, hipMemcpyDeviceToDevice, stream);
  hipMemcpyAsync(dhf,  dec_h0, (size_t)BB * HD * 4,     hipMemcpyDeviceToDevice, stream);
  hipMemcpyAsync(dcf,  dec_c0, (size_t)BB * HD * 4,     hipMemcpyDeviceToDevice, stream);
  int n;
  n = 4 * BB * HE;  k_cvt_bf16<<<(n + 255) / 256, 256, 0, stream>>>(hstf, hstb, n);
  n = BB * HD;      k_cvt_bf16<<<(n + 255) / 256, 256, 0, stream>>>(dhf, dhb, n);
  n = BB * SS * FF; k_cvt_bf16<<<(n + 255) / 256, 256, 0, stream>>>(x, xbf, n);

  // ---- pack weights [Wih|Whh] -> bf16 (fwd/bwd contiguous per layer) ----
  k_packw<<<G4, 128, 0, stream>>>(eW0f, eU0f, eb0f, ec0f, FF, wc_e0, bc);
  k_packw<<<G4, 128, 0, stream>>>(eW0b, eU0b, eb0b, ec0b, FF,
                                  wc_e0 + (size_t)G4 * (FF + HE), bc + G4);
  k_packw<<<G4, 128, 0, stream>>>(eW1f, eU1f, eb1f, ec1f, EO, wc_e1, bc + 2 * G4);
  k_packw<<<G4, 128, 0, stream>>>(eW1b, eU1b, eb1b, ec1b, EO,
                                  wc_e1 + (size_t)G4 * (EO + HE), bc + 3 * G4);
  k_packw<<<G4, 128, 0, stream>>>(dW, dU, db, dcb, EO + FF, wc_d, bc + 4 * G4);

  // ---- encoder layer 0 (fwd+bwd concurrently, gridDim.z = 2) ----
  dim3 ggrid(G4 / 64, BB / 128, 2);
  dim3 ugrid(BB * HE / 256, 2);
  for (int t = 0; t < SS; ++t) {
    k_lstm_gemm<<<ggrid, 128, 0, stream>>>(xbf, FF, SS * FF, hstb, BB * HE,
                                           wc_e0, G4 * (FF + HE), bc, gbuf, t, 1);
    k_lstm_update<<<ugrid, 256, 0, stream>>>(gbuf, hstf, cstf, hstb, BB * HE,
                                             inp1, EO, HE, t, 1);
  }
  // ---- encoder layer 1 ----
  for (int t = 0; t < SS; ++t) {
    k_lstm_gemm<<<ggrid, 128, 0, stream>>>(inp1, EO, SS * EO, hstb + 2 * BB * HE,
                                           BB * HE, wc_e1, G4 * (EO + HE),
                                           bc + 2 * G4, gbuf, t, 1);
    k_lstm_update<<<ugrid, 256, 0, stream>>>(gbuf, hstf + 2 * BB * HE,
                                             cstf + 2 * BB * HE,
                                             hstb + 2 * BB * HE, BB * HE,
                                             encout, EO, HE, t, 1);
  }

  // ---- attention ----
  k_decterm<<<BB, 128, 0, stream>>>(dec_h0, aW, decterm);
  k_scores<<<BB * SS, 128, 0, stream>>>(encout, aW, ab, decterm, scores);
  k_softmax<<<BB, 256, 0, stream>>>(scores, alphaw, out_alpha);
  k_build_decin<<<BB * SS, 128, 0, stream>>>(alphaw, encout, xbf, decin);

  // ---- decoder LSTM ----
  dim3 ggridd(G4 / 64, BB / 128, 1);
  dim3 ugridd(BB * HD / 256, 1);
  for (int t = 0; t < SS; ++t) {
    k_lstm_gemm<<<ggridd, 128, 0, stream>>>(decin, EO + FF, SS * (EO + FF), dhb, 0,
                                            wc_d, 0, bc + 4 * G4, gbuf, t, 0);
    k_lstm_update<<<ugridd, 256, 0, stream>>>(gbuf, dhf, dcf, dhb, 0,
                                              decout, HD, 0, t, 0);
  }

  // ---- FC head ----
  k_fc1<<<dim3(8, BB), 256, 0, stream>>>(decout, f1W, f1b, h1);
  k_out<<<1, 256, 0, stream>>>(h1, f2W, f2b, out);

  // ---- final state outputs ----
  hipMemcpyAsync(out_hs, hstf, (size_t)4 * BB * HE * 4, hipMemcpyDeviceToDevice, stream);
  hipMemcpyAsync(out_cs, cstf, (size_t)4 * BB * HE * 4, hipMemcpyDeviceToDevice, stream);
  hipMemcpyAsync(out_dh, dhf,  (size_t)BB * HD * 4,     hipMemcpyDeviceToDevice, stream);
  hipMemcpyAsync(out_dc, dcf,  (size_t)BB * HD * 4,     hipMemcpyDeviceToDevice, stream);
}
